// SpatialGNN_10565619548639
// MI455X (gfx1250) — matrix operand order repaired and checked by reading, not compiled, not run on hardware
//
#include <hip/hip_runtime.h>

typedef __attribute__((ext_vector_type(16))) _Float16 v16h;
typedef __attribute__((ext_vector_type(8)))  float    v8f;

#define TPB 256

// ---------- ordered-float encoding for atomic max over floats ----------
__device__ __forceinline__ unsigned f32_to_ord(float f) {
  unsigned u = __float_as_uint(f);
  return (u & 0x80000000u) ? ~u : (u | 0x80000000u);
}
__device__ __forceinline__ float ord_to_f32(unsigned u) {
  return (u & 0x80000000u) ? __uint_as_float(u & 0x7fffffffu)
                           : __uint_as_float(~u);
}
#define ORD_NEG_INF 0x007FFFFFu  // f32_to_ord(-inf)

// ---------- fills / converts ----------
__global__ void fill_f32_k(float* p, float v, long n) {
  long t = (long)blockIdx.x * blockDim.x + threadIdx.x;
  if (t < n) p[t] = v;
}
__global__ void fill_u32_k(unsigned* p, unsigned v, long n) {
  long t = (long)blockIdx.x * blockDim.x + threadIdx.x;
  if (t < n) p[t] = v;
}
__global__ void f32_to_f16_k(const float* __restrict__ a, _Float16* __restrict__ b, int n) {
  int t = blockIdx.x * blockDim.x + threadIdx.x;
  if (t < n) b[t] = (_Float16)a[t];
}

// ---------- WMMA GEMM: C[M,Nc] = A[M,K](f32) * Bh[K,Nc](f16) ----------
// One 16x16 output tile per wave; K stepped by 32 with v_wmma_f32_16x16x32_f16.
// Fragment packing follows CDNA5 ISA 7.12.2 (wave32 layouts).
__global__ __launch_bounds__(TPB)
void gemm_f16w_k(const float* __restrict__ A, const _Float16* __restrict__ Bh,
                 float* __restrict__ C, int M, int K, int Nc) {
  const int gwave = (blockIdx.x * blockDim.x + threadIdx.x) >> 5;
  const int lane  = threadIdx.x & 31;
  const int ntiles = Nc >> 4;
  const int total  = (M >> 4) * ntiles;
  if (gwave >= total) return;
  const int tm  = gwave / ntiles;
  const int tn  = gwave - tm * ntiles;
  const int l15 = lane & 15;
  const int g   = lane >> 4;             // half-wave group
  const int row = (tm << 4) + l15;       // A row for this lane
  const int col = (tn << 4) + l15;       // B/C column for this lane

  v8f acc = {};
  for (int kb = 0; kb < K; kb += 32) {
    // A fragment 16x32 f16: lanes0-15 hold K {kb+0..7, kb+16..23},
    // lanes16-31 hold K {kb+8..15, kb+24..31}
    v16h a;
    const float* ap0 = A + (size_t)row * K + kb + 8 * g;
    const float* ap1 = ap0 + 16;
#pragma unroll
    for (int i = 0; i < 8; ++i) {
      a[i]     = (_Float16)ap0[i];
      a[8 + i] = (_Float16)ap1[i];
    }
    // B fragment 32x16 f16: lane holds column `col`; elements 0..7 are
    // K = kb + 8*g + 0..7, elements 8..15 are K = kb + 16 + 8*g + 0..7
    v16h b;
    const _Float16* bp = Bh + (size_t)(kb + 8 * g) * Nc + col;
#pragma unroll
    for (int i = 0; i < 8; ++i) {
      b[i]     = bp[(size_t)i * Nc];
      b[8 + i] = bp[(size_t)(16 + i) * Nc];
    }

    acc = __builtin_amdgcn_wmma_f32_16x16x32_f16(
        /*neg_a=*/false, a, /*neg_b=*/false, b,
        /*c_mod=*/(short)0, acc, /*reuse_a=*/false, /*reuse_b=*/false);
  }
  // C/D layout: VGPR r -> M = 8*g + r, N = col
  const int m0 = (tm << 4) + 8 * g;
#pragma unroll
  for (int r = 0; r < 8; ++r) C[(size_t)(m0 + r) * Nc + col] = acc[r];
}

// ---------- attention scores: es/ed[n,h] = <h[n,h,:], a_src/dst[h,:]> ----------
__global__ void att_scores_k(const float* __restrict__ h, const float* __restrict__ asrc,
                             const float* __restrict__ adst, float* __restrict__ es,
                             float* __restrict__ ed, int n, int heads, int d) {
  int t = blockIdx.x * blockDim.x + threadIdx.x;
  if (t >= n * heads) return;
  int node = t / heads, hh = t - node * heads;
  const float* hp = h + (size_t)node * heads * d + hh * d;
  const float* as = asrc + hh * d;
  const float* ad = adst + hh * d;
  float s = 0.f, dd = 0.f;
  for (int o = 0; o < d; ++o) { float v = hp[o]; s += v * as[o]; dd += v * ad[o]; }
  es[t] = s; ed[t] = dd;
}

__device__ __forceinline__ void edge_nodes(const int* __restrict__ srcs,
                                           const int* __restrict__ dsts,
                                           int E, int eid, int& s, int& d) {
  if (eid < E) { s = srcs[eid]; d = dsts[eid]; }
  else         { s = eid - E; d = eid - E; }   // appended self-loops
}

// ---------- pass 1: e = leaky_relu(es[src]+ed[dst]); segment max over dst ----------
__global__ void edge_max_k(const int* __restrict__ srcs, const int* __restrict__ dsts,
                           const float* __restrict__ es, const float* __restrict__ ed,
                           float* __restrict__ ebuf, unsigned* __restrict__ dmax,
                           int E, int EP, int heads) {
  int t = blockIdx.x * blockDim.x + threadIdx.x;
  if (t >= EP * heads) return;
  int eid = t / heads, hh = t - eid * heads;
  int s, d; edge_nodes(srcs, dsts, E, eid, s, d);
  float e = es[(size_t)s * heads + hh] + ed[(size_t)d * heads + hh];
  e = e > 0.f ? e : 0.2f * e;            // leaky_relu, slope 0.2
  ebuf[t] = e;
  atomicMax(&dmax[(size_t)d * heads + hh], f32_to_ord(e));
}

// ---------- pass 2: a = exp(e - max[dst]); segment sum over dst ----------
__global__ void edge_exp_k(const int* __restrict__ srcs, const int* __restrict__ dsts,
                           float* __restrict__ ebuf, const unsigned* __restrict__ dmax,
                           float* __restrict__ denom, int E, int EP, int heads) {
  int t = blockIdx.x * blockDim.x + threadIdx.x;
  if (t >= EP * heads) return;
  int eid = t / heads, hh = t - eid * heads;
  int s, d; edge_nodes(srcs, dsts, E, eid, s, d);
  float m = ord_to_f32(dmax[(size_t)d * heads + hh]);
  float a = __expf(ebuf[t] - m);
  ebuf[t] = a;
  atomicAdd(&denom[(size_t)d * heads + hh], a);
}

// ---------- pass 3: agg[dst] += h[src] * (a / denom[dst]) ----------
// one thread per (edge, feature) -> coalesced h[src] reads within a wave
__global__ void edge_msg_k(const int* __restrict__ srcs, const int* __restrict__ dsts,
                           const float* __restrict__ ebuf, const float* __restrict__ denom,
                           const float* __restrict__ h, float* __restrict__ agg,
                           int E, long total, int heads, int d) {
  long t = (long)blockIdx.x * blockDim.x + threadIdx.x;
  if (t >= total) return;
  const int c = heads * d;
  int feat = (int)(t % c);
  int eid  = (int)(t / c);
  int hh   = feat / d;
  int s, dd; edge_nodes(srcs, dsts, E, eid, s, dd);
  float alpha = ebuf[(size_t)eid * heads + hh] / denom[(size_t)dd * heads + hh];
  atomicAdd(&agg[(size_t)dd * c + feat], h[(size_t)s * c + feat] * alpha);
}

// ---------- epilogue: x += bias (broadcast), optional relu ----------
__global__ void bias_act_k(float* __restrict__ x, const float* __restrict__ b,
                           long n, int c, int relu) {
  long t = (long)blockIdx.x * blockDim.x + threadIdx.x;
  if (t >= n * (long)c) return;
  float v = x[t] + b[(int)(t % c)];
  if (relu) v = v > 0.f ? v : 0.f;
  x[t] = v;
}

// ---------- global mean pool (accumulate) ----------
__global__ void pool_k(const float* __restrict__ hf, const int* __restrict__ batch,
                       float* __restrict__ pooled, float* __restrict__ cnt, int n, int c) {
  long t = (long)blockIdx.x * blockDim.x + threadIdx.x;
  if (t >= (long)n * c) return;
  int node = (int)(t / c);
  int o    = (int)(t - (long)node * c);
  int g = batch[node];
  atomicAdd(&pooled[(size_t)g * c + o], hf[t]);
  if (o == 0) atomicAdd(&cnt[g], 1.0f);
}
__global__ void pool_div_k(float* pooled, const float* cnt, int G, int c) {
  int t = blockIdx.x * blockDim.x + threadIdx.x;
  if (t >= G * c) return;
  pooled[t] /= fmaxf(cnt[t / c], 1.0f);
}

// ---------- tiny MLP layer: Y[m,n] = act(X[m,k] @ W[k,n] + b) ----------
__global__ void mlp_k(const float* __restrict__ X, const float* __restrict__ W,
                      const float* __restrict__ b, float* __restrict__ Y,
                      int m, int k, int n, int relu) {
  int t = blockIdx.x * blockDim.x + threadIdx.x;
  if (t >= m * n) return;
  int r = t / n, cc = t - r * n;
  float s = b[cc];
  for (int i = 0; i < k; ++i) s += X[(size_t)r * k + i] * W[(size_t)i * n + cc];
  if (relu) s = fmaxf(s, 0.f);
  Y[t] = s;
}

extern "C" void kernel_launch(void* const* d_in, const int* in_sizes, int n_in,
                              void* d_out, int out_size, void* d_ws, size_t ws_size,
                              hipStream_t stream) {
  const float* x   = (const float*)d_in[0];
  const int*   ei  = (const int*)d_in[1];
  const int*   bat = (const int*)d_in[2];
  const float* W1  = (const float*)d_in[3];
  const float* as1 = (const float*)d_in[4];
  const float* ad1 = (const float*)d_in[5];
  const float* b1  = (const float*)d_in[6];
  const float* W2  = (const float*)d_in[7];
  const float* as2 = (const float*)d_in[8];
  const float* ad2 = (const float*)d_in[9];
  const float* b2  = (const float*)d_in[10];
  const float* dw1 = (const float*)d_in[11];
  const float* db1 = (const float*)d_in[12];
  const float* dw2 = (const float*)d_in[13];
  const float* db2 = (const float*)d_in[14];
  float* out = (float*)d_out;
  (void)n_in; (void)out_size; (void)ws_size;

  const int IN = 128, HID = 64, OUTF = 16, G = 64;
  const int n  = in_sizes[0] / IN;     // nodes
  const int E  = in_sizes[1] / 2;      // edges (before self-loops)
  const int EP = E + n;                // with self-loops
  const int C1 = 2 * HID;              // layer-1 concat width = 128
  const int* srcs = ei;
  const int* dsts = ei + E;

  // ---- carve workspace ----
  char* w = (char*)d_ws;
  size_t off = 0;
  auto alloc = [&](size_t bytes) -> void* {
    void* p = w + off;
    off = (off + bytes + 255) & ~(size_t)255;
    return p;
  };
  _Float16* Wh1   = (_Float16*)alloc((size_t)IN * C1 * 2);
  _Float16* Wh2   = (_Float16*)alloc((size_t)C1 * HID * 2);
  float*    hbuf  = (float*)alloc((size_t)n * C1 * 4);   // h1, reused as h2
  float*    agg   = (float*)alloc((size_t)n * C1 * 4);   // agg1 -> x2 -> agg2
  float*    es    = (float*)alloc((size_t)n * 2 * 4);
  float*    ed    = (float*)alloc((size_t)n * 2 * 4);
  unsigned* dmax  = (unsigned*)alloc((size_t)n * 2 * 4);
  float*    denom = (float*)alloc((size_t)n * 2 * 4);
  float*    ebuf  = (float*)alloc((size_t)EP * 2 * 4);
  float*    pooled= (float*)alloc((size_t)G * HID * 4);
  float*    cnt   = (float*)alloc((size_t)G * 4);
  float*    z     = (float*)alloc((size_t)G * HID * 4);

  auto nb = [](long total, int tpb) { return (int)((total + tpb - 1) / tpb); };

  // ---- weights to f16 ----
  f32_to_f16_k<<<nb(IN * C1, TPB), TPB, 0, stream>>>(W1, Wh1, IN * C1);
  f32_to_f16_k<<<nb(C1 * HID, TPB), TPB, 0, stream>>>(W2, Wh2, C1 * HID);

  // ================= Layer 1 (heads=2, d=64) =================
  {
    long waves = (long)(n / 16) * (C1 / 16);
    gemm_f16w_k<<<nb(waves * 32, TPB), TPB, 0, stream>>>(x, Wh1, hbuf, n, IN, C1);
  }
  att_scores_k<<<nb((long)n * 2, TPB), TPB, 0, stream>>>(hbuf, as1, ad1, es, ed, n, 2, HID);
  fill_u32_k<<<nb((long)n * 2, TPB), TPB, 0, stream>>>(dmax, ORD_NEG_INF, (long)n * 2);
  fill_f32_k<<<nb((long)n * 2, TPB), TPB, 0, stream>>>(denom, 0.f, (long)n * 2);
  fill_f32_k<<<nb((long)n * C1, TPB), TPB, 0, stream>>>(agg, 0.f, (long)n * C1);
  edge_max_k<<<nb((long)EP * 2, TPB), TPB, 0, stream>>>(srcs, dsts, es, ed, ebuf, dmax, E, EP, 2);
  edge_exp_k<<<nb((long)EP * 2, TPB), TPB, 0, stream>>>(srcs, dsts, ebuf, dmax, denom, E, EP, 2);
  edge_msg_k<<<nb((long)EP * C1, TPB), TPB, 0, stream>>>(srcs, dsts, ebuf, denom, hbuf, agg,
                                                         E, (long)EP * C1, 2, HID);
  bias_act_k<<<nb((long)n * C1, TPB), TPB, 0, stream>>>(agg, b1, (long)n, C1, 1); // relu -> x2

  // ================= Layer 2 (heads=1, d=64) =================
  {
    long waves = (long)(n / 16) * (HID / 16);
    gemm_f16w_k<<<nb(waves * 32, TPB), TPB, 0, stream>>>(agg, Wh2, hbuf, n, C1, HID);
  }
  att_scores_k<<<nb((long)n, TPB), TPB, 0, stream>>>(hbuf, as2, ad2, es, ed, n, 1, HID);
  fill_u32_k<<<nb((long)n, TPB), TPB, 0, stream>>>(dmax, ORD_NEG_INF, (long)n);
  fill_f32_k<<<nb((long)n, TPB), TPB, 0, stream>>>(denom, 0.f, (long)n);
  fill_f32_k<<<nb((long)n * HID, TPB), TPB, 0, stream>>>(agg, 0.f, (long)n * HID); // after GEMM read
  edge_max_k<<<nb((long)EP, TPB), TPB, 0, stream>>>(srcs, dsts, es, ed, ebuf, dmax, E, EP, 1);
  edge_exp_k<<<nb((long)EP, TPB), TPB, 0, stream>>>(srcs, dsts, ebuf, dmax, denom, E, EP, 1);
  edge_msg_k<<<nb((long)EP * HID, TPB), TPB, 0, stream>>>(srcs, dsts, ebuf, denom, hbuf, agg,
                                                          E, (long)EP * HID, 1, HID);
  bias_act_k<<<nb((long)n * HID, TPB), TPB, 0, stream>>>(agg, b2, (long)n, HID, 0);

  // ================= mean pool + decoder MLP =================
  fill_f32_k<<<nb(G * HID, TPB), TPB, 0, stream>>>(pooled, 0.f, G * HID);
  fill_f32_k<<<nb(G, TPB), TPB, 0, stream>>>(cnt, 0.f, G);
  pool_k<<<nb((long)n * HID, TPB), TPB, 0, stream>>>(agg, bat, pooled, cnt, n, HID);
  pool_div_k<<<nb(G * HID, TPB), TPB, 0, stream>>>(pooled, cnt, G, HID);
  mlp_k<<<nb(G * HID, TPB), TPB, 0, stream>>>(pooled, dw1, db1, z, G, HID, HID, 1);
  mlp_k<<<nb(G * OUTF, TPB), TPB, 0, stream>>>(z, dw2, db2, out, G, HID, OUTF, 0);
}
